// HyperTriplet_40243843564012
// MI455X (gfx1250) — matrix operand order repaired and checked by reading, not compiled, not run on hardware
//
#include <hip/hip_runtime.h>
#include <hip/hip_bf16.h>
#include <math.h>

// ---------------------------------------------------------------------------
// Types for CDNA5 WMMA
// ---------------------------------------------------------------------------
typedef _Float16 half8  __attribute__((ext_vector_type(8)));
typedef _Float16 half16 __attribute__((ext_vector_type(16)));
typedef float    float8 __attribute__((ext_vector_type(8)));

#define MAXNORM_C 0.996f       // (1 - 4e-3)/sqrt(c), c=1
#define ATANH_EPS 1e-5f

// gfx1250 async global->LDS copy path (ASYNCcnt tracked). Set to 0 to fall
// back to synchronous global_load + ds_store staging.
#define USE_ASYNC_LDS 1

__device__ __forceinline__ float artanh_clip(float x) {
    x = fminf(fmaxf(x, -1.f + ATANH_EPS), 1.f - ATANH_EPS);
    return 0.5f * logf((1.f + x) / (1.f - x));
}

__device__ __forceinline__ float block_sum(float v, float* sd) {
    const int t = threadIdx.x;
    sd[t] = v;
    __syncthreads();
#pragma unroll
    for (int s = 128; s > 0; s >>= 1) {
        if (t < s) sd[t] += sd[t + s];
        __syncthreads();
    }
    float r = sd[0];
    __syncthreads();
    return r;
}

// ---------------------------------------------------------------------------
// Stage 0: per-block z-score + concat + expmap0 -> f16 activations (zero-pad)
// One block per row. Up to 4 feature segments (pass w=0 to skip).
// ---------------------------------------------------------------------------
__global__ __launch_bounds__(256) void zscore_expmap0_kernel(
    const float* __restrict__ s0, int w0,
    const float* __restrict__ s1, int w1,
    const float* __restrict__ s2, int w2,
    const float* __restrict__ s3, int w3,
    _Float16* __restrict__ act, float* __restrict__ norm_out, int Kpad)
{
    __shared__ float buf[1728];
    __shared__ float sd[256];
    const int row = blockIdx.x, tid = threadIdx.x;
    const float* segs[4] = { s0, s1, s2, s3 };
    const int    wid[4]  = { w0, w1, w2, w3 };

    int offset = 0;
    for (int sg = 0; sg < 4; ++sg) {
        const float* p = segs[sg];
        const int    w = wid[sg];
        if (p == nullptr || w == 0) continue;
        float ps = 0.f, pq = 0.f;
        for (int i = tid; i < w; i += 256) {
            float x = p[(size_t)row * w + i];
            ps += x; pq += x * x;
        }
        float s = block_sum(ps, sd);
        float q = block_sum(pq, sd);
        float mean = s / (float)w;
        float var  = (q - (float)w * mean * mean) / (float)(w - 1);  // ddof=1
        float stdv = sqrtf(fmaxf(var, 0.f)) + 1e-8f;
        for (int i = tid; i < w; i += 256)
            buf[offset + i] = (p[(size_t)row * w + i] - mean) / stdv;
        offset += w;
        __syncthreads();
    }

    // expmap0: tanh(|u|) * u / |u|
    float pn = 0.f;
    for (int i = tid; i < offset; i += 256) { float u = buf[i]; pn += u * u; }
    float n = fmaxf(sqrtf(fmaxf(block_sum(pn, sd), 1e-30f)), 1e-15f);
    float scale = tanhf(n) / n;
    for (int i = tid; i < Kpad; i += 256) {
        float v = (i < offset) ? buf[i] * scale : 0.f;
        act[(size_t)row * Kpad + i] = (_Float16)v;
    }
    if (tid == 0) norm_out[row] = fmaxf(tanhf(n), 1e-15f);
}

// ---------------------------------------------------------------------------
// Weight prep: W f32 [K,N] row-major -> Wt f16 [Npad,Kpad] (transposed, padded)
// ---------------------------------------------------------------------------
__global__ void convert_w_kernel(const float* __restrict__ W, _Float16* __restrict__ Wt,
                                 int K, int N, int Kpad, int Npad)
{
    int idx = blockIdx.x * 256 + threadIdx.x;
    int total = Npad * Kpad;
    if (idx >= total) return;
    int n = idx / Kpad;
    int k = idx - n * Kpad;
    float v = (n < N && k < K) ? W[(size_t)k * N + n] : 0.f;
    Wt[idx] = (_Float16)v;
}

// ---------------------------------------------------------------------------
// WMMA GEMM: C[B,N] (f32) = A[B,Kpad] (f16) x Wt[Npad,Kpad]^T (f16)
// Block: 256 threads (8 wave32s). Tile: BM=256, BN=64, BK=32.
// Wave w computes rows [w*32, w*32+32) x 4 N-fragments of 16: 8 WMMAs per
// K-step, each B fragment reused by two v_wmma_f32_16x16x32_f16.
// Global->LDS staging uses double-buffered GLOBAL_LOAD_ASYNC_TO_LDS_B128
// (ASYNCcnt) so the copy of tile k+1 overlaps the WMMAs on tile k.
// ---------------------------------------------------------------------------
#define BM 256
#define BN 64
#define BK 32
#define SA 40   // padded LDS row stride in halfs (80B) to spread banks

__global__ __launch_bounds__(256) void gemm_f16_wmma(
    const _Float16* __restrict__ A, const _Float16* __restrict__ Wt,
    float* __restrict__ C, int N, int Kpad)
{
    __shared__ _Float16 As[2][BM * SA];
    __shared__ _Float16 Bs[2][BN * SA];

    const int tid  = threadIdx.x;
    const int m0   = blockIdx.x * BM;
    const int n0   = blockIdx.y * BN;
    const int wave = tid >> 5;
    const int lane = tid & 31;
    const int lm   = lane & 15;   // 0..15
    const int lh   = lane >> 4;   // 0 or 1

    // staging map: A -> one row per thread (32 halfs = 4 x b128);
    //              B -> 4 threads per row (8 halfs = 1 x b128 each)
    const _Float16* aSrc = A  + (size_t)(m0 + tid) * Kpad;
    const _Float16* bSrc = Wt + (size_t)(n0 + (tid >> 2)) * Kpad + (tid & 3) * 8;
    const int aDst = tid * SA;
    const int bDst = (tid >> 2) * SA + (tid & 3) * 8;

    float8 acc[2][4] = {};

    auto issue_tile = [&](int k0, int b) {
#if USE_ASYNC_LDS
        unsigned la = (unsigned)(size_t)&As[b][aDst];
        unsigned long long ga = (unsigned long long)(size_t)(aSrc + k0);
#pragma unroll
        for (int s = 0; s < 4; ++s) {
            asm volatile("global_load_async_to_lds_b128 %0, %1, off"
                         :: "v"(la + s * 16), "v"(ga + s * 16) : "memory");
        }
        unsigned lb = (unsigned)(size_t)&Bs[b][bDst];
        unsigned long long gb = (unsigned long long)(size_t)(bSrc + k0);
        asm volatile("global_load_async_to_lds_b128 %0, %1, off"
                     :: "v"(lb), "v"(gb) : "memory");
#else
        const _Float16* ga = aSrc + k0;
        _Float16* da = &As[b][aDst];
#pragma unroll
        for (int s = 0; s < 4; ++s)
            *reinterpret_cast<half8*>(da + s * 8) =
                *reinterpret_cast<const half8*>(ga + s * 8);
        *reinterpret_cast<half8*>(&Bs[b][bDst]) =
            *reinterpret_cast<const half8*>(bSrc + k0);
#endif
    };

    issue_tile(0, 0);
    int buf = 0;

    for (int k0 = 0; k0 < Kpad; k0 += BK) {
#if USE_ASYNC_LDS
        asm volatile("s_wait_asynccnt 0x0" ::: "memory");
#endif
        __syncthreads();   // tile `buf` fully in LDS; all reads of tile buf^1 done
        if (k0 + BK < Kpad) issue_tile(k0 + BK, buf ^ 1);

        // ---- A fragments (16x32 f16 each): lanes 0-15 -> K 0..7 / 16..23,
        //      lanes 16-31 -> K 8..15 / 24..31  (ISA 7.12.2)
        half16 af[2];
#pragma unroll
        for (int mh = 0; mh < 2; ++mh) {
            const _Float16* ap = &As[buf][(wave * 32 + mh * 16 + lm) * SA + lh * 8];
            half8 lo = *reinterpret_cast<const half8*>(ap);
            half8 hi = *reinterpret_cast<const half8*>(ap + 16);
#pragma unroll
            for (int i = 0; i < 8; ++i) { af[mh][i] = lo[i]; af[mh][i + 8] = hi[i]; }
        }

        // ---- 4 B fragments, each feeding two WMMAs. B (32x16): lanes 0-15 ->
        //      K 0..7 / 16..23, lanes 16-31 -> K 8..15 / 24..31, N = lane%16.
#pragma unroll
        for (int j = 0; j < 4; ++j) {
            const _Float16* bp = &Bs[buf][(j * 16 + lm) * SA + lh * 8];
            half8 lo = *reinterpret_cast<const half8*>(bp);
            half8 hi = *reinterpret_cast<const half8*>(bp + 16);
            half16 bf;
#pragma unroll
            for (int i = 0; i < 8; ++i) { bf[i] = lo[i]; bf[i + 8] = hi[i]; }
            acc[0][j] = __builtin_amdgcn_wmma_f32_16x16x32_f16(
                false, af[0], false, bf, (short)0, acc[0][j], false, false);
            acc[1][j] = __builtin_amdgcn_wmma_f32_16x16x32_f16(
                false, af[1], false, bf, (short)0, acc[1][j], false, false);
        }
        buf ^= 1;
    }

    // ---- store D: VGPR r -> M = r (lanes 0-15) / r+8 (lanes 16-31), N = lane%16
#pragma unroll
    for (int mh = 0; mh < 2; ++mh) {
        const int mbase = m0 + wave * 32 + mh * 16 + lh * 8;
#pragma unroll
        for (int j = 0; j < 4; ++j) {
            const int n = n0 + j * 16 + lm;
            if (n < N) {
#pragma unroll
                for (int r = 0; r < 8; ++r)
                    C[(size_t)(mbase + r) * N + n] = acc[mh][j][r];
            }
        }
    }
}

// ---------------------------------------------------------------------------
// Fused hyperbolic-linear epilogue (one block per row, N <= 1024):
//   mobius_matvec scalar part -> projx -> mobius_add(bias) -> projx
//   -> optional mobius_tanh (logmap0, tanh, expmap0, projx)
// Emits next-layer f16 activations (zero-padded) and/or f32 output,
// plus the exact row norm consumed by the next layer.
// ---------------------------------------------------------------------------
__global__ __launch_bounds__(256) void hlinear_epilogue(
    const float* __restrict__ mx, const float* __restrict__ bias,
    const float* __restrict__ norm_in, int N,
    _Float16* __restrict__ act_out, int Kpad_next,
    float* __restrict__ f32_out, float* __restrict__ norm_out, int do_tanh)
{
    __shared__ float sd[256];
    const int row = blockIdx.x, tid = threadIdx.x;
    const float* m = mx + (size_t)row * N;

    float vv[4], bb[4];
    // ||mx||
    float p = 0.f;
#pragma unroll
    for (int e = 0; e < 4; ++e) {
        int i = tid + e * 256;
        float v = (i < N) ? m[i] : 0.f;
        vv[e] = v; p += v * v;
    }
    float mxn = fmaxf(sqrtf(fmaxf(block_sum(p, sd), 1e-30f)), 1e-15f);

    // mobius_matvec: y = tanh((||mx||/||x||) * artanh(||x||)) * mx/||mx||
    float xn = norm_in[row];
    float r  = (mxn / xn) * artanh_clip(xn);
    float ty = tanhf(r);
    float yn = fmaxf(fabsf(ty), 1e-15f);
    float sy = ty / mxn;
    // projx (norm known analytically)
    float pscale = (yn > MAXNORM_C) ? (MAXNORM_C / yn) : 1.f;
    yn = fminf(yn, MAXNORM_C);
    sy *= pscale;

    // mobius_add(y, b): need y.b and |b|^2
    float pd = 0.f, pb = 0.f;
#pragma unroll
    for (int e = 0; e < 4; ++e) {
        int i = tid + e * 256;
        if (i < N) {
            float bi = bias[i];
            float y  = sy * vv[e];
            bb[e] = bi; vv[e] = y;
            pd += y * bi; pb += bi * bi;
        } else { bb[e] = 0.f; }
    }
    float xy = block_sum(pd, sd);
    float b2 = block_sum(pb, sd);
    float y2 = yn * yn;
    float c1 = 1.f + 2.f * xy + b2;
    float c2 = 1.f - y2;
    float den = fmaxf(1.f + 2.f * xy + y2 * b2, 1e-15f);

    float pz = 0.f;
#pragma unroll
    for (int e = 0; e < 4; ++e) {
        int i = tid + e * 256;
        if (i < N) {
            float z = (c1 * vv[e] + c2 * bb[e]) / den;
            vv[e] = z; pz += z * z;
        }
    }
    float zn = fmaxf(sqrtf(fmaxf(block_sum(pz, sd), 1e-30f)), 1e-15f);
    float pj = (zn > MAXNORM_C) ? (MAXNORM_C / zn) : 1.f;

    float fn;
    if (do_tanh) {
        // mobius_tanh: projx(expmap0(tanh(logmap0(z))))
        float zc = fminf(zn, MAXNORM_C);
        float f  = artanh_clip(zc) / zc;
        float pt = 0.f;
#pragma unroll
        for (int e = 0; e < 4; ++e) {
            int i = tid + e * 256;
            if (i < N) {
                float t = tanhf(f * pj * vv[e]);
                vv[e] = t; pt += t * t;
            }
        }
        float tn = fmaxf(sqrtf(fmaxf(block_sum(pt, sd), 1e-30f)), 1e-15f);
        float wn = tanhf(tn);
        float g  = wn / tn;
        float pw = (wn > MAXNORM_C) ? (MAXNORM_C / wn) : 1.f;
#pragma unroll
        for (int e = 0; e < 4; ++e) vv[e] *= g * pw;
        fn = fminf(wn, MAXNORM_C);
    } else {
#pragma unroll
        for (int e = 0; e < 4; ++e) vv[e] *= pj;
        fn = fminf(zn, MAXNORM_C);
    }
    fn = fmaxf(fn, 1e-15f);

    if (act_out) {
        _Float16* ao = act_out + (size_t)row * Kpad_next;
#pragma unroll
        for (int e = 0; e < 4; ++e) {
            int i = tid + e * 256;
            if (i < Kpad_next) ao[i] = (_Float16)((i < N) ? vv[e] : 0.f);
        }
    }
    if (f32_out) {
        float* fo = f32_out + (size_t)row * N;
#pragma unroll
        for (int e = 0; e < 4; ++e) {
            int i = tid + e * 256;
            if (i < N) fo[i] = vv[e];
        }
    }
    if (tid == 0) norm_out[row] = fn;
}

// ---------------------------------------------------------------------------
// mid = projx(mobius_add(mobius_scalar_mul(0.27, head), mobius_scalar_mul(0.73, tail)))
// HD == 256 == blockDim.x; one block per row.
// ---------------------------------------------------------------------------
__global__ __launch_bounds__(256) void mid_combine(
    const float* __restrict__ head, const float* __restrict__ tail,
    const float* __restrict__ nh, const float* __restrict__ nt,
    _Float16* __restrict__ act, float* __restrict__ norm_out)
{
    __shared__ float sd[256];
    const int row = blockIdx.x, tid = threadIdx.x;
    float h  = head[(size_t)row * 256 + tid];
    float t  = tail[(size_t)row * 256 + tid];
    float hn = nh[row], tn = nt[row];

    float ta = tanhf(0.27f * artanh_clip(hn));
    float tb = tanhf(0.73f * artanh_clip(tn));
    float a  = (ta / hn) * h;
    float b  = (tb / tn) * t;
    float an = fmaxf(fabsf(ta), 1e-15f);
    float bn = fmaxf(fabsf(tb), 1e-15f);

    float xy = block_sum(a * b, sd);
    float a2 = an * an, b2 = bn * bn;
    float den = fmaxf(1.f + 2.f * xy + a2 * b2, 1e-15f);
    float z = ((1.f + 2.f * xy + b2) * a + (1.f - a2) * b) / den;

    float zn = fmaxf(sqrtf(fmaxf(block_sum(z * z, sd), 1e-30f)), 1e-15f);
    float pj = (zn > MAXNORM_C) ? (MAXNORM_C / zn) : 1.f;
    act[(size_t)row * 256 + tid] = (_Float16)(z * pj);
    if (tid == 0) norm_out[row] = fmaxf(fminf(zn, MAXNORM_C), 1e-15f);
}

// ---------------------------------------------------------------------------
// Host side
// ---------------------------------------------------------------------------
extern "C" void kernel_launch(void* const* d_in, const int* in_sizes, int n_in,
                              void* d_out, int out_size, void* d_ws, size_t ws_size,
                              hipStream_t stream)
{
    (void)n_in; (void)out_size; (void)ws_size;
    const int Bsz = in_sizes[0] / 768;   // 16384

    const float* dcb  = (const float*)d_in[0];
    const float* dpc  = (const float*)d_in[1];
    const float* dmac = (const float*)d_in[2];
    const float* decf = (const float*)d_in[3];
    const float* tesm = (const float*)d_in[4];
    const float* tpc  = (const float*)d_in[5];
    const float* tmer = (const float*)d_in[6];
    const float* Wd1 = (const float*)d_in[7];  const float* bd1 = (const float*)d_in[8];
    const float* Wd2 = (const float*)d_in[9];  const float* bd2 = (const float*)d_in[10];
    const float* Wd3 = (const float*)d_in[11]; const float* bd3 = (const float*)d_in[12];
    const float* WT1 = (const float*)d_in[13]; const float* bT1 = (const float*)d_in[14];
    const float* WT2 = (const float*)d_in[15]; const float* bT2 = (const float*)d_in[16];
    const float* WT3 = (const float*)d_in[17]; const float* bT3 = (const float*)d_in[18];
    const float* WM1 = (const float*)d_in[19]; const float* bM1 = (const float*)d_in[20];
    const float* WM2 = (const float*)d_in[21]; const float* bM2 = (const float*)d_in[22];
    const float* WM3 = (const float*)d_in[23]; const float* bM3 = (const float*)d_in[24];

    // workspace carve
    char* ws = (char*)d_ws;
    size_t off = 0;
    auto carve = [&](size_t bytes) -> char* {
        char* p = ws + off;
        off += (bytes + 255) & ~(size_t)255;
        return p;
    };
    _Float16* actA  = (_Float16*)carve((size_t)Bsz * 1728 * sizeof(_Float16));
    _Float16* actB  = (_Float16*)carve((size_t)Bsz * 1024 * sizeof(_Float16));
    float*    mxbuf = (float*)   carve((size_t)Bsz * 1024 * sizeof(float));
    float*    headF = (float*)   carve((size_t)Bsz * 256  * sizeof(float));
    float*    tailF = (float*)   carve((size_t)Bsz * 256  * sizeof(float));
    _Float16* wtbuf = (_Float16*)carve((size_t)1024 * 1728 * sizeof(_Float16));
    float* nrm0 = (float*)carve((size_t)Bsz * sizeof(float));
    float* nrm1 = (float*)carve((size_t)Bsz * sizeof(float));
    float* nrmH = (float*)carve((size_t)Bsz * sizeof(float));
    float* nrmT = (float*)carve((size_t)Bsz * sizeof(float));
    float* nrmM = (float*)carve((size_t)Bsz * sizeof(float));

    auto layer = [&](const float* W, const float* bias, int K, int Kpad, int N,
                     const _Float16* aIn, const float* nIn,
                     _Float16* aOut, float* fOut, float* nOut, int do_tanh) {
        const int Npad = (N + 63) & ~63;
        const int tot  = Npad * Kpad;
        convert_w_kernel<<<(tot + 255) / 256, 256, 0, stream>>>(W, wtbuf, K, N, Kpad, Npad);
        gemm_f16_wmma<<<dim3(Bsz / BM, Npad / BN), 256, 0, stream>>>(aIn, wtbuf, mxbuf, N, Kpad);
        hlinear_epilogue<<<Bsz, 256, 0, stream>>>(mxbuf, bias, nIn, N, aOut, N, fOut, nOut, do_tanh);
    };

    // ---- drug branch ----
    zscore_expmap0_kernel<<<Bsz, 256, 0, stream>>>(dcb, 768, dpc, 11, dmac, 167, decf, 765,
                                                   actA, nrm0, 1728);
    layer(Wd1, bd1, 1711, 1728, 1024, actA, nrm0, actB, nullptr, nrm1, 1);
    layer(Wd2, bd2, 1024, 1024, 512,  actB, nrm1, actA, nullptr, nrm0, 1);
    layer(Wd3, bd3, 512,  512,  256,  actA, nrm0, nullptr, headF, nrmH, 0);

    // ---- target branch ----
    zscore_expmap0_kernel<<<Bsz, 256, 0, stream>>>(tesm, 1280, tpc, 11, tmer, 420, nullptr, 0,
                                                   actA, nrm0, 1728);
    layer(WT1, bT1, 1711, 1728, 1024, actA, nrm0, actB, nullptr, nrm1, 1);
    layer(WT2, bT2, 1024, 1024, 512,  actB, nrm1, actA, nullptr, nrm0, 1);
    layer(WT3, bT3, 512,  512,  256,  actA, nrm0, nullptr, tailF, nrmT, 0);

    // ---- combine + mid head ----
    mid_combine<<<Bsz, 256, 0, stream>>>(headF, tailF, nrmH, nrmT, actA, nrmM);
    layer(WM1, bM1, 256, 256, 64, actA, nrmM, actB, nullptr, nrm1, 1);
    layer(WM2, bM2, 64,  64,  32, actB, nrm1, actA, nullptr, nrm0, 1);
    layer(WM3, bM3, 32,  32,  2,  actA, nrm0, nullptr, (float*)d_out, nrm1, 0);
}
